// CapsLayerWithClusterRouting_55894704390186
// MI455X (gfx1250) — matrix operand order repaired and checked by reading, not compiled, not run on hardware
//
#include <hip/hip_runtime.h>
#include <math.h>

typedef __attribute__((ext_vector_type(16))) __bf16 v16bf;
typedef __attribute__((ext_vector_type(8)))  float  v8f;

#define APITCH 168                          // bf16 elems per A row in LDS (336B, 16B aligned)
#define WR_BYTES (4096u*160u*2u)            // 1,310,720
#define CAPS_BYTES (16u*8u*16u*64u*64u*4u)  // 33,554,432
#define BSTAGE (64*160)                     // bf16 elems per B stage (4 k-tiles x 16 rows x 160)

union Frag { uint4 q[2]; v16bf v; };

__device__ __forceinline__ void async_b128(unsigned long long gaddr, unsigned ldsoff) {
  asm volatile("global_load_async_to_lds_b128 %0, %1, off"
               :: "v"(ldsoff), "v"(gaddr) : "memory");
}
__device__ __forceinline__ void wait_async0() {
  asm volatile("s_wait_asynccnt 0x0" ::: "memory");
}
__device__ __forceinline__ void wait_ds0() {
  asm volatile("s_wait_dscnt 0x0" ::: "memory");
}

// ---------------- prep: repack weights to bf16 [co][t*16+c] (pad 144->160), zero LN sums
__global__ void caps_prep(const float* __restrict__ W, __bf16* __restrict__ Wr,
                          float* __restrict__ sums) {
  int idx = blockIdx.x * 256 + threadIdx.x;          // 4096*160 = 655360 total
  if (idx < 4096 * 160) {
    int co = idx / 160, kv = idx - co * 160;
    float v = 0.f;
    if (kv < 144) {
      int t = kv >> 4, c = kv & 15;                  // W: [co][c][kh][kw] = co*144 + c*9 + t
      v = W[co * 144 + c * 9 + t];
    }
    Wr[idx] = (__bf16)v;
  }
  if (blockIdx.x == 0 && threadIdx.x < 256) sums[threadIdx.x] = 0.f; // 128 sum + 128 sumsq
}

// ---------------- conv (implicit GEMM, bf16 WMMA, async-staged shared weights) + routing
__global__ void __launch_bounds__(256) caps_conv_route(
    const float* __restrict__ X, const __bf16* __restrict__ Wr,
    const float* __restrict__ bias, float* __restrict__ capsbuf,
    float* __restrict__ sums)
{
  __shared__ __align__(16) __bf16 Alds[8 * 16 * APITCH];   // 8 wave tiles x 16 pos x 168
  __shared__ __align__(16) __bf16 Bbuf[2 * BSTAGE];        // double-buffered weight stage
  __shared__ float Scr[8 * 16 * 17];                       // per-wave transpose scratch

  int blk = blockIdx.x;                 // 512 = b(16) x hpair(32); WG covers 2 rows x 64 w
  int h0 = (blk & 31) << 1;
  int b  = blk >> 5;
  int tid = threadIdx.x;

  // Build all 8 A tiles: tile t -> (h = h0 + t/4, w0 = (t%4)*16); k = tap*16 + cin
  for (int e = 0; e < 80; ++e) {
    int idx = tid * 80 + e;             // 20480 elements
    int t   = idx / 2560;
    int rem = idx - t * 2560;
    int p   = rem / 160;
    int kv  = rem - p * 160;
    float v = 0.f;
    if (kv < 144) {
      int tap = kv >> 4, c = kv & 15;
      int kh = tap / 3;
      int hh = h0 + (t >> 2) + kh - 1;
      int ww = ((t & 3) << 4) + p + (tap - kh * 3) - 1;
      if (hh >= 0 && hh < 64 && ww >= 0 && ww < 64)
        v = X[((b * 16 + c) * 64 + hh) * 64 + ww];
    }
    Alds[(t * 16 + p) * APITCH + kv] = (__bf16)v;
  }

  int lane = tid & 31, wid = tid >> 5;  // wave id == spatial tile
  int l16 = lane & 15, half = lane >> 4;

  // Per-thread weight-staging constants: thread owns 80B of one weight row per stage.
  int row_lin = tid >> 2, part = tid & 3;     // row_lin = kblk*16 + r
  int kblk = row_lin >> 4, r = row_lin & 15;
  unsigned long long gW = (unsigned long long)(uintptr_t)Wr
      + (unsigned long long)((kblk * 128 + r) * 160 + part * 40) * 2ull;
  unsigned ldsW = (unsigned)(uintptr_t)&Bbuf[0] + (unsigned)(row_lin * 320 + part * 80);

  __syncthreads();                       // A tiles visible

  // A fragments: invariant across all 256 channel tiles (shared reduction axis)
  Frag afrag[5];
  {
    const __bf16* arow = &Alds[(wid * 16 + l16) * APITCH];
    #pragma unroll
    for (int ks = 0; ks < 5; ++ks) {
      afrag[ks].q[0] = *(const uint4*)(arow + ks * 32 + half * 8);
      afrag[ks].q[1] = *(const uint4*)(arow + ks * 32 + 16 + half * 8);
    }
  }

  // Prologue: async-stage weights for stage 0 (c_out=0, ci=0) into Bbuf[0]
  #pragma unroll
  for (int q = 0; q < 5; ++q) async_b128(gW + q * 16, ldsW + q * 16);

  v8f amax, se, wsum;
  for (int s = 0; s < 64; ++s) {         // stage s = c_out*8 + ci
    int c_out = s >> 3, ci = s & 7;
    if (ci == 0) {
      #pragma unroll
      for (int i = 0; i < 8; ++i) { amax[i] = -__builtin_inff(); se[i] = 0.f; wsum[i] = 0.f; }
    }
    wait_async0();                       // stage s weights landed in LDS
    __syncthreads();                     // ... for every wave; prev stage fully consumed
    if (s < 63) {                        // overlap: DMA stage s+1 while computing stage s
      int sn = s + 1;
      unsigned long long ga = gW
          + (unsigned long long)(((sn & 7) * 512 + (sn >> 3) * 16) * 160) * 2ull;
      unsigned lo = ldsW + (unsigned)((sn & 1) * (BSTAGE * 2));
      #pragma unroll
      for (int q = 0; q < 5; ++q) async_b128(ga + q * 16, lo + q * 16);
    }

    const __bf16* bb = &Bbuf[(s & 1) * BSTAGE];
    v8f sumv, sqv;
    #pragma unroll
    for (int i = 0; i < 8; ++i) { sumv[i] = 0.f; sqv[i] = 0.f; }
    #pragma unroll
    for (int k = 0; k < 4; ++k) {
      const __bf16* brow = bb + (k * 16 + l16) * 160 + half * 8;
      v8f acc = {};
      #pragma unroll
      for (int ks = 0; ks < 5; ++ks) {
        Frag bf_;
        bf_.q[0] = *(const uint4*)(brow + ks * 32);
        bf_.q[1] = *(const uint4*)(brow + ks * 32 + 16);
        acc = __builtin_amdgcn_wmma_f32_16x16x32_bf16(
            false, afrag[ks].v, false, bf_.v, (short)0, acc, false, false);
      }
      float bv = bias[ci * 512 + k * 128 + c_out * 16 + l16];
      #pragma unroll
      for (int i = 0; i < 8; ++i) {
        float x = acc[i] + bv;
        sumv[i] += x; sqv[i] += x * x;
      }
    }
    // vote mean/var over K=4, agreement = -log(std), online softmax over ci
    #pragma unroll
    for (int i = 0; i < 8; ++i) {
      float mean = sumv[i] * 0.25f;
      float var  = fmaxf(sqv[i] * 0.25f - mean * mean, 1e-38f);
      float ag = -0.5f * logf(var);
      float mn = fmaxf(amax[i], ag);
      float sc = expf(amax[i] - mn);
      float e  = expf(ag - mn);
      se[i]   = se[i] * sc + e;
      wsum[i] = wsum[i] * sc + e * mean;
      amax[i] = mn;
    }

    if (ci == 7) {                       // finalize this c_out for this wave's tile
      float s1 = 0.f, s2 = 0.f;
      float cval[8];
      #pragma unroll
      for (int i = 0; i < 8; ++i) {
        float c = wsum[i] / se[i];
        cval[i] = c; s1 += c; s2 += c * c;
      }
      #pragma unroll
      for (int off = 16; off >= 1; off >>= 1) {
        s1 += __shfl_xor(s1, off, 32);
        s2 += __shfl_xor(s2, off, 32);
      }
      if (lane == 0) {
        atomicAdd(&sums[b * 8 + c_out], s1);
        atomicAdd(&sums[128 + b * 8 + c_out], s2);
      }
      // transpose (M=pos, N=d) -> coalesced [d][w] stores; in-wave LDS, DS is in-order
      float* scr = &Scr[wid * (16 * 17)];
      #pragma unroll
      for (int i = 0; i < 8; ++i)
        scr[(i + half * 8) * 17 + l16] = cval[i];     // scr[pos][d]
      wait_ds0();
      int h_w = h0 + (wid >> 2);
      int w0_w = (wid & 3) << 4;
      float* cbase = capsbuf + ((size_t)(b * 8 + c_out) * 16) * 4096 + h_w * 64 + w0_w;
      #pragma unroll
      for (int j = 0; j < 8; ++j) {
        int d = j + half * 8;
        cbase[d * 4096 + l16] = scr[l16 * 17 + d];
      }
    }
  }
}

// ---------------- finalize: LayerNorm over (D,h,w) per (b,c_out), write (C_out,B,D,h,w)
__global__ void caps_finalize(const float4* __restrict__ capsbuf,
                              const float* __restrict__ sums,
                              const float4* __restrict__ gamma,
                              const float4* __restrict__ beta,
                              float4* __restrict__ out)
{
  int g = blockIdx.x * 256 + threadIdx.x;  // float4 index over output, 2,097,152 total
  int co   = g >> 18;
  int rem  = g & 262143;
  int b    = rem >> 14;
  int rem2 = rem & 16383;
  int d    = rem2 >> 10;
  int hw4  = rem2 & 1023;
  float sum = sums[b * 8 + co];
  float sq  = sums[128 + b * 8 + co];
  const float inv_n = 1.0f / 65536.0f;
  float mean = sum * inv_n;
  float var  = sq * inv_n - mean * mean;
  float rstd = rsqrtf(var + 1e-5f);
  float4 cv = capsbuf[((b * 8 + co) * 16 + d) * 1024 + hw4];
  float4 gm = gamma[d * 1024 + hw4];
  float4 bt = beta[d * 1024 + hw4];
  float4 o;
  o.x = (cv.x - mean) * rstd * gm.x + bt.x;
  o.y = (cv.y - mean) * rstd * gm.y + bt.y;
  o.z = (cv.z - mean) * rstd * gm.z + bt.z;
  o.w = (cv.w - mean) * rstd * gm.w + bt.w;
  out[g] = o;
}

extern "C" void kernel_launch(void* const* d_in, const int* in_sizes, int n_in,
                              void* d_out, int out_size, void* d_ws, size_t ws_size,
                              hipStream_t stream) {
  const float* caps_in = (const float*)d_in[0];  // (16,16,64,64)
  const float* W       = (const float*)d_in[1];  // (4096,16,3,3)
  const float* bias    = (const float*)d_in[2];  // (4096,)
  const float* gamma   = (const float*)d_in[3];  // (16,64,64)
  const float* beta    = (const float*)d_in[4];  // (16,64,64)

  char* ws = (char*)d_ws;
  __bf16* Wr      = (__bf16*)ws;                          // 1,310,720 B
  float*  capsbuf = (float*)(ws + WR_BYTES);              // 33,554,432 B
  float*  sums    = (float*)(ws + WR_BYTES + CAPS_BYTES); // 256 floats

  caps_prep<<<2560, 256, 0, stream>>>(W, Wr, sums);
  caps_conv_route<<<512, 256, 0, stream>>>(caps_in, Wr, bias, capsbuf, sums);
  caps_finalize<<<8192, 256, 0, stream>>>((const float4*)capsbuf, sums,
                                          (const float4*)gamma, (const float4*)beta,
                                          (float4*)d_out);
}
